// GatingLayer_3719441679101
// MI455X (gfx1250) — matrix operand order repaired and checked by reading, not compiled, not run on hardware
//
#include <hip/hip_runtime.h>
#include <hip/hip_bf16.h>
#include <cstddef>

typedef __attribute__((ext_vector_type(16))) __bf16 v16bf;
typedef __attribute__((ext_vector_type(8)))  float  v8f;

// Problem constants (fixed by the reference harness)
#define BATCH 32
#define CIN   512
#define COUT  1024
#define HDIM  56
#define WDIM  56
#define HW    3136   // 56*56

// ---------------------------------------------------------------------------
// Kernel 1: x = W[1024x512] @ In_b[512x3136] + bias (per batch), bf16 WMMA.
// Block: 256 threads = 8 waves as 4(M) x 2(N). Each wave: 32M x 64N... no:
// each wave computes a 32M x 32N macro-tile (2 A-frags x 2 B-frags = 4 WMMA
// per 32-K step). Block tile: 128M x 64N. Grid: (3136/64, 1024/128, BATCH).
// LDS B-tile is double-buffered: one barrier per k-step.
// ---------------------------------------------------------------------------
__global__ __launch_bounds__(256) void gemm_proj_kernel(
    const float* __restrict__ input,   // [B, CIN, HW]
    const float* __restrict__ w_proj,  // [COUT, CIN] row-major
    const float* __restrict__ b_proj,  // [COUT]
    float* __restrict__ xbuf)          // [B, COUT, HW]
{
    const int nt   = blockIdx.x;           // 0..48  (64 N cols each)
    const int mb   = blockIdx.y;           // 0..7   (128 M rows each)
    const int b    = blockIdx.z;           // batch
    const int tid  = threadIdx.x;
    const int lane = tid & 31;
    const int wave = tid >> 5;             // 0..7
    const int mw   = wave >> 1;            // 0..3  (32-row slice)
    const int nw   = wave & 1;             // 0..1  (32-col slice)

    // B fragments, wave32 layout, 4 n-subtiles x [32 lanes][16 bf16], x2 bufs
    __shared__ __align__(32) __bf16 sFrag[2][4 * 512];

    const float* inb = input + (size_t)b * CIN * HW;
    const int n0  = nt * 64;
    const int kg  = lane >> 4;             // A/D K-half / M-half select
    const int l15 = lane & 15;
    const int mrow0 = mb * 128 + mw * 32 + l15;   // A rows (subtile 0)

    // ---- stage 32K x 64N fp32 tile -> bf16 LDS in B-fragment order --------
    auto stage = [&](int k0, int buf) {
        #pragma unroll
        for (int i = 0; i < 8; ++i) {
            const int idx = tid + (i << 8);        // 0..2047, no tail
            const int kk  = idx >> 6;              // 0..31
            const int nn  = idx & 63;              // 0..63
            const int sub = nn >> 4;               // n-subtile 0..3
            const int dl  = (nn & 15) + (((kk >> 3) & 1) << 4);  // frag lane
            const int e   = (kk & 7) + ((kk >> 4) << 3);  // frag element
            sFrag[buf][sub * 512 + dl * 16 + e] =
                (__bf16)inb[(size_t)(k0 + kk) * HW + (n0 + nn)];
        }
    };

    // ---- A fragment: row `row`, K = k0 + kg*8 + [0..7] and +16 ------------
    auto loadA = [&](int row, int k0) -> v16bf {
        const float4* w4 =
            reinterpret_cast<const float4*>(w_proj + (size_t)row * CIN + k0 + kg * 8);
        const float4 fa0 = w4[0], fa1 = w4[1], fb0 = w4[4], fb1 = w4[5];
        v16bf a;
        a[0]  = (__bf16)fa0.x; a[1]  = (__bf16)fa0.y;
        a[2]  = (__bf16)fa0.z; a[3]  = (__bf16)fa0.w;
        a[4]  = (__bf16)fa1.x; a[5]  = (__bf16)fa1.y;
        a[6]  = (__bf16)fa1.z; a[7]  = (__bf16)fa1.w;
        a[8]  = (__bf16)fb0.x; a[9]  = (__bf16)fb0.y;
        a[10] = (__bf16)fb0.z; a[11] = (__bf16)fb0.w;
        a[12] = (__bf16)fb1.x; a[13] = (__bf16)fb1.y;
        a[14] = (__bf16)fb1.z; a[15] = (__bf16)fb1.w;
        return a;
    };

    v8f acc00 = {}, acc01 = {}, acc10 = {}, acc11 = {};

    stage(0, 0);

    #pragma unroll
    for (int ks = 0; ks < CIN / 32; ++ks) {
        const int k0  = ks * 32;
        const int cur = ks & 1;
        __syncthreads();                       // staging of `cur` complete;
                                               // reads of `cur^1` (prev iter) done
        if (ks + 1 < CIN / 32) stage(k0 + 32, cur ^ 1);

        const v16bf a0 = loadA(mrow0,      k0);
        const v16bf a1 = loadA(mrow0 + 16, k0);
        const v16bf b0 = *reinterpret_cast<const v16bf*>(
            &sFrag[cur][(nw * 2 + 0) * 512 + lane * 16]);
        const v16bf b1 = *reinterpret_cast<const v16bf*>(
            &sFrag[cur][(nw * 2 + 1) * 512 + lane * 16]);

        acc00 = __builtin_amdgcn_wmma_f32_16x16x32_bf16(false, a0, false, b0,
                                                        (short)0, acc00, false, false);
        acc01 = __builtin_amdgcn_wmma_f32_16x16x32_bf16(false, a0, false, b1,
                                                        (short)0, acc01, false, false);
        acc10 = __builtin_amdgcn_wmma_f32_16x16x32_bf16(false, a1, false, b0,
                                                        (short)0, acc10, false, false);
        acc11 = __builtin_amdgcn_wmma_f32_16x16x32_bf16(false, a1, false, b1,
                                                        (short)0, acc11, false, false);
    }

    // ---- store D (+bias). C/D f32 layout: vgpr r -> M=r+8*(lane>=16), N=lane&15
    float* outb = xbuf + (size_t)b * COUT * HW;
    const int mbase = mb * 128 + mw * 32;
    const int ng0 = n0 + (nw * 2 + 0) * 16 + l15;
    const int ng1 = n0 + (nw * 2 + 1) * 16 + l15;
    #pragma unroll
    for (int r = 0; r < 8; ++r) {
        const int m0 = mbase + r + (kg << 3);        // A-subtile 0 rows
        const int m1 = mbase + 16 + r + (kg << 3);   // A-subtile 1 rows
        const float bia0 = b_proj[m0];
        const float bia1 = b_proj[m1];
        outb[(size_t)m0 * HW + ng0] = acc00[r] + bia0;
        outb[(size_t)m0 * HW + ng1] = acc01[r] + bia0;
        outb[(size_t)m1 * HW + ng0] = acc10[r] + bia1;
        outb[(size_t)m1 * HW + ng1] = acc11[r] + bia1;
    }
}

// ---------------------------------------------------------------------------
// Kernel 2: per (b,c): discounted cumsum over H then W on x_scan, then
// out = input + gamma * x_gate * sigmoid(X_cum). One 64-thread block / tile.
// ---------------------------------------------------------------------------
__global__ __launch_bounds__(64) void scan_gate_kernel(
    const float* __restrict__ input,     // [B, CIN, HW]
    const float* __restrict__ xbuf,      // [B, COUT, HW]
    const float* __restrict__ discount,  // [CIN]
    const float* __restrict__ gamma,     // [1]
    float* __restrict__ out)             // [B, CIN, HW]
{
    const int bc  = blockIdx.x;
    const int b   = bc >> 9;     // / 512
    const int c   = bc & 511;
    const int tid = threadIdx.x;

    __shared__ float tile[HW];

    const float* xs = xbuf  + ((size_t)b * COUT + CIN + c) * HW;  // x_scan
    const float* xg = xbuf  + ((size_t)b * COUT + c) * HW;        // x_gate
    const float* ip = input + ((size_t)b * CIN  + c) * HW;
    float*       op = out   + ((size_t)b * CIN  + c) * HW;

    const float d = discount[c];
    const float g = gamma[0];

    for (int i = tid; i < HW; i += 64) tile[i] = xs[i];
    __syncthreads();

    // Scan along H: one column per thread (lanes consecutive -> no conflicts)
    if (tid < WDIM) {
        float a = 0.f;
        #pragma unroll 4
        for (int h = 0; h < HDIM; ++h) {
            a = d * a + tile[h * WDIM + tid];
            tile[h * WDIM + tid] = a;
        }
    }
    __syncthreads();

    // Scan along W: one row per thread
    if (tid < HDIM) {
        float a = 0.f;
        #pragma unroll 4
        for (int w = 0; w < WDIM; ++w) {
            a = d * a + tile[tid * WDIM + w];
            tile[tid * WDIM + w] = a;
        }
    }
    __syncthreads();

    // Gate + residual, coalesced
    for (int i = tid; i < HW; i += 64) {
        const float s = 1.f / (1.f + __expf(-tile[i]));
        op[i] = ip[i] + g * (xg[i] * s);
    }
}

// ---------------------------------------------------------------------------
extern "C" void kernel_launch(void* const* d_in, const int* in_sizes, int n_in,
                              void* d_out, int out_size, void* d_ws, size_t ws_size,
                              hipStream_t stream)
{
    (void)in_sizes; (void)n_in; (void)out_size; (void)ws_size;
    const float* input    = (const float*)d_in[0];
    const float* w_proj   = (const float*)d_in[1];
    const float* b_proj   = (const float*)d_in[2];
    const float* discount = (const float*)d_in[3];
    const float* gamma    = (const float*)d_in[4];
    float* xbuf = (float*)d_ws;                 // needs 32*1024*3136*4 B ~ 411 MB
    float* outp = (float*)d_out;

    dim3 g1(HW / 64, COUT / 128, BATCH);        // (49, 8, 32)
    gemm_proj_kernel<<<g1, 256, 0, stream>>>(input, w_proj, b_proj, xbuf);

    scan_gate_kernel<<<BATCH * CIN, 64, 0, stream>>>(input, xbuf, discount,
                                                     gamma, outp);
}
